// se3_denoiser_87857851007397
// MI455X (gfx1250) — matrix operand order repaired and checked by reading, not compiled, not run on hardware
//
#include <hip/hip_runtime.h>
#include <math.h>
#include <stdint.h>

#define N_NODES 65536
#define IN_CH   512
#define HID     256
#define OUT_CH  8
#define NUM_GRAPHS 32
#define MAX_LEN 4096

// LDS staging geometry for the GEMM B-tile: 256 cols x 32 k, padded to 40 elems/col.
#define BCOLS   256
#define BSTRIDE 40                       // bf16 elems per col (80B, 16B-aligned, 20-bank stride)
#define BBUFB   (BCOLS * BSTRIDE * 2)    // bytes per buffer = 20480

typedef __attribute__((ext_vector_type(16))) __bf16 v16bf;
typedef __attribute__((ext_vector_type(8)))  __bf16 v8bf;
typedef __attribute__((ext_vector_type(8)))  float  v8f;

__device__ __forceinline__ unsigned short f2bf(float f) {
    unsigned int u = __float_as_uint(f);
    unsigned int r = (u + 0x7FFFu + ((u >> 16) & 1u)) >> 16;   // round-nearest-even
    return (unsigned short)r;
}
__device__ __forceinline__ float bf2f(unsigned short h) {
    return __uint_as_float(((unsigned int)h) << 16);
}
__device__ __forceinline__ float gelu_exact(float x) {
    return 0.5f * x * (1.0f + erff(x * 0.70710678118654752f));
}
__device__ __forceinline__ v16bf cat16(v8bf lo, v8bf hi) {
    return __builtin_shufflevector(lo, hi, 0,1,2,3,4,5,6,7,8,9,10,11,12,13,14,15);
}
__device__ __forceinline__ void wait_async0() {
#if __has_builtin(__builtin_amdgcn_s_wait_asynccnt)
    __builtin_amdgcn_s_wait_asynccnt(0);
#else
    asm volatile("s_wait_asynccnt 0x0" ::: "memory");
#endif
}

// Issue 64B (4 x b128) of one B column chunk into LDS via the async-to-LDS path (ASYNCcnt).
__device__ __forceinline__ void stage_B_async(const unsigned short* __restrict__ Bt,
                                              int K, int nblk, int tid, int kb,
                                              unsigned ldsDst) {
    unsigned long long ga =
        (unsigned long long)(uintptr_t)(Bt + (size_t)(nblk + tid) * K + (size_t)kb * 32);
#pragma unroll
    for (int j = 0; j < 4; ++j) {
        unsigned la = ldsDst + j * 16;
        unsigned long long g = ga + (unsigned long long)(j * 16);
        asm volatile("global_load_async_to_lds_b128 %0, %1, off"
                     :: "v"(la), "v"(g) : "memory");
    }
}

// ---------------- BN statistics ----------------
__global__ void k_zero(float* p, int n) {
    int i = blockIdx.x * blockDim.x + threadIdx.x;
    if (i < n) p[i] = 0.0f;
}

// 256 blocks x 256 threads; each block reduces 256 rows, each thread handles 2 channels.
__global__ void k_bn_partial(const float* __restrict__ X, float* __restrict__ sums) {
    int t  = threadIdx.x;                 // channel t and t+256
    int r0 = blockIdx.x * 256;
    const float* base = X + (size_t)r0 * IN_CH;
    float s0 = 0.f, q0 = 0.f, s1 = 0.f, q1 = 0.f;
    for (int r = 0; r < 256; ++r) {
        float a = base[(size_t)r * IN_CH + t];
        float b = base[(size_t)r * IN_CH + 256 + t];
        s0 += a; q0 += a * a;
        s1 += b; q1 += b * b;
    }
    atomicAdd(&sums[t],           s0);
    atomicAdd(&sums[t + 256],     s1);
    atomicAdd(&sums[512 + t],     q0);
    atomicAdd(&sums[512 + t+256], q1);
}

__global__ void k_bn_finalize(const float* __restrict__ sums,
                              const float* __restrict__ gamma,
                              const float* __restrict__ beta,
                              float* __restrict__ ss) {
    int c = blockIdx.x * blockDim.x + threadIdx.x;
    if (c < IN_CH) {
        const float inv = 1.0f / (float)N_NODES;
        float mu  = sums[c] * inv;
        float var = sums[512 + c] * inv - mu * mu;
        float sc  = gamma[c] * rsqrtf(var + 1e-5f);
        ss[c]         = sc;
        ss[IN_CH + c] = beta[c] - mu * sc;
    }
}

// normalize + convert to bf16, 4 elements/thread
__global__ void k_bn_apply_cvt(const float* __restrict__ X, const float* __restrict__ ss,
                               unsigned short* __restrict__ Xn) {
    size_t i4 = ((size_t)blockIdx.x * blockDim.x + threadIdx.x) * 4;
    const float4 x = *reinterpret_cast<const float4*>(X + i4);
    int c = (int)(i4 & (IN_CH - 1));
    float v0 = x.x * ss[c]     + ss[IN_CH + c];
    float v1 = x.y * ss[c + 1] + ss[IN_CH + c + 1];
    float v2 = x.z * ss[c + 2] + ss[IN_CH + c + 2];
    float v3 = x.w * ss[c + 3] + ss[IN_CH + c + 3];
    uint2 pk;
    pk.x = (unsigned)f2bf(v0) | ((unsigned)f2bf(v1) << 16);
    pk.y = (unsigned)f2bf(v2) | ((unsigned)f2bf(v3) << 16);
    *reinterpret_cast<uint2*>(Xn + i4) = pk;
}

// Wt[n*K + k] = bf16(W[k*N + n])   (coalesced writes)
__global__ void k_transpose_cvt(const float* __restrict__ W, unsigned short* __restrict__ Wt,
                                int K, int N) {
    int idx = blockIdx.x * blockDim.x + threadIdx.x;
    if (idx < K * N) {
        int n = idx / K, k = idx - n * K;
        Wt[(size_t)n * K + k] = f2bf(W[(size_t)k * N + n]);
    }
}

// ---------------- bf16 WMMA GEMM: C[MxN] = gelu?(A[MxK] * Bt[NxK]^T + bias) ----------------
// Wave tile 32x64 (2 M-subtiles x 4 N-subtiles, v_wmma_f32_16x16x32_bf16).
// Block = 256 threads = 8 waves arranged 2(M) x 4(N) -> 64x256 block tile.
// B chunk (256 cols x 32 k) is double-buffered in LDS via global_load_async_to_lds_b128.
__global__ __launch_bounds__(256) void k_gemm_bf16(
    const unsigned short* __restrict__ A,
    const unsigned short* __restrict__ Bt,
    const float* __restrict__ bias,
    unsigned short* __restrict__ C,
    int K, int N, int doGelu)
{
    __shared__ unsigned short sB[2][BCOLS][BSTRIDE];

    const int tid   = threadIdx.x;
    const int lane  = tid & 31;
    const int wave  = tid >> 5;
    const int lmod  = lane & 15;
    const int lhalf = lane >> 4;            // 0 or 1
    const int wm = wave >> 2;               // 0..1
    const int wn = wave & 3;                // 0..3
    const int mbase = blockIdx.x * 64 + wm * 32;
    const int nblk  = blockIdx.y * 256;     // first column this block covers

    const unsigned sB0 = (unsigned)(unsigned long long)(uintptr_t)&sB[0][0][0];
    const unsigned myDst0 = sB0 + (unsigned)(tid * (BSTRIDE * 2));          // buffer 0 slot
    const unsigned myDst1 = myDst0 + (unsigned)BBUFB;                       // buffer 1 slot

    v8f acc[2][4];
    v8f zero = {};
#pragma unroll
    for (int t = 0; t < 2; ++t)
#pragma unroll
        for (int nt = 0; nt < 4; ++nt) acc[t][nt] = zero;

    const unsigned short* arow0 = A + (size_t)(mbase + lmod) * K;
    const unsigned short* arow1 = A + (size_t)(mbase + 16 + lmod) * K;

    const int ksteps = K >> 5;

    // prologue: stage k-chunk 0 into buffer 0
    stage_B_async(Bt, K, nblk, tid, 0, myDst0);
    wait_async0();
    __syncthreads();

    int buf = 0;
    for (int kb = 0; kb < ksteps; ++kb) {
        // overlap: issue next B chunk into the other buffer
        if (kb + 1 < ksteps)
            stage_B_async(Bt, K, nblk, tid, kb + 1, buf ? myDst0 : myDst1);

        const int k0 = kb * 32;
        __builtin_prefetch(arow0 + k0 + 32, 0, 1);
        __builtin_prefetch(arow1 + k0 + 32, 0, 1);
        // A layout (16-bit 16x32): lanes 0-15 hold K{0..7,16..23}, lanes 16-31 K{8..15,24..31}
        v8bf a0lo = *(const v8bf*)(arow0 + k0 + lhalf * 8);
        v8bf a0hi = *(const v8bf*)(arow0 + k0 + 16 + lhalf * 8);
        v8bf a1lo = *(const v8bf*)(arow1 + k0 + lhalf * 8);
        v8bf a1hi = *(const v8bf*)(arow1 + k0 + 16 + lhalf * 8);
        v16bf a0 = cat16(a0lo, a0hi);
        v16bf a1 = cat16(a1lo, a1hi);
#pragma unroll
        for (int nt = 0; nt < 4; ++nt) {
            // B layout (16-bit 32x16): lane n<16 = col n K{0..7,16..23}, lane n+16 = col n K{8..15,24..31}
            const int coll = wn * 64 + nt * 16 + lmod;
            const v8bf* bp = reinterpret_cast<const v8bf*>(&sB[buf][coll][lhalf * 8]);
            v8bf blo = bp[0];
            v8bf bhi = bp[2];
            v16bf b = cat16(blo, bhi);
            acc[0][nt] = __builtin_amdgcn_wmma_f32_16x16x32_bf16(
                false, a0, false, b, (short)0, acc[0][nt], false, false);
            acc[1][nt] = __builtin_amdgcn_wmma_f32_16x16x32_bf16(
                false, a1, false, b, (short)0, acc[1][nt], false, false);
        }
        // wait for next chunk's async writes; barrier also ensures everyone is
        // done reading the buffer the following iteration's issue overwrites
        wait_async0();
        __syncthreads();
        buf ^= 1;
    }

    // epilogue: C/D layout: lane -> col = lane%16, rows = (lane/16)*8 + j
#pragma unroll
    for (int t = 0; t < 2; ++t) {
#pragma unroll
        for (int nt = 0; nt < 4; ++nt) {
            int n = nblk + wn * 64 + nt * 16 + lmod;
            float bn = bias[n];
#pragma unroll
            for (int j = 0; j < 8; ++j) {
                int m = mbase + t * 16 + lhalf * 8 + j;
                float v = acc[t][nt][j] + bn;
                if (doGelu) v = gelu_exact(v);
                C[(size_t)m * N + n] = f2bf(v);
            }
        }
    }
}

// ---------------- ragged bookkeeping + head ----------------
__global__ void k_offsets(const int* __restrict__ batch, int* __restrict__ offs) {
    int i = blockIdx.x * blockDim.x + threadIdx.x;
    if (i >= N_NODES) return;
    if (i == 0) offs[batch[0]] = 0;
    else if (batch[i] != batch[i - 1]) offs[batch[i]] = i;
}

// pad value = tanh(dyt(gelu(0 @ Wo + bo))) for empty slots
__global__ void k_fill_out(float* __restrict__ out,
                           const float* __restrict__ bo,
                           const float* __restrict__ alpha,
                           const float* __restrict__ dw,
                           const float* __restrict__ db) {
    size_t i = (size_t)blockIdx.x * blockDim.x + threadIdx.x;
    int c = (int)(i & (OUT_CH - 1));
    float a = gelu_exact(bo[c]);
    a = dw[c] * tanhf(alpha[0] * a) + db[c];
    out[i] = tanhf(a);
}

__global__ __launch_bounds__(256) void k_head(
    const unsigned short* __restrict__ H3,
    const float* __restrict__ Wo, const float* __restrict__ bo,
    const int* __restrict__ batch, const int* __restrict__ offs,
    const float* __restrict__ alpha, const float* __restrict__ dw,
    const float* __restrict__ db, float* __restrict__ out)
{
    __shared__ float sWo[HID * OUT_CH];
    for (int j = threadIdx.x; j < HID * OUT_CH; j += 256) sWo[j] = Wo[j];
    __syncthreads();

    int i = blockIdx.x * 256 + threadIdx.x;
    float acc[OUT_CH];
#pragma unroll
    for (int c = 0; c < OUT_CH; ++c) acc[c] = 0.0f;

    const unsigned short* h = H3 + (size_t)i * HID;
#pragma unroll 4
    for (int k = 0; k < HID; ++k) {
        float hv = bf2f(h[k]);
#pragma unroll
        for (int c = 0; c < OUT_CH; ++c) acc[c] += hv * sWo[k * OUT_CH + c];
    }

    int g   = batch[i];
    int pos = i - offs[g];
    float al = alpha[0];
    size_t o = ((size_t)pos * NUM_GRAPHS + g) * OUT_CH;
#pragma unroll
    for (int c = 0; c < OUT_CH; ++c) {
        float y = gelu_exact(acc[c] + bo[c]);
        y = dw[c] * tanhf(al * y) + db[c];
        out[o + c] = tanhf(y);
    }
}

// ---------------- launch ----------------
extern "C" void kernel_launch(void* const* d_in, const int* in_sizes, int n_in,
                              void* d_out, int out_size, void* d_ws, size_t ws_size,
                              hipStream_t stream) {
    const float* x     = (const float*)d_in[0];
    const int*   batch = (const int*)  d_in[1];
    const float* gam   = (const float*)d_in[2];
    const float* bet   = (const float*)d_in[3];
    const float* W1    = (const float*)d_in[4];
    const float* b1    = (const float*)d_in[5];
    const float* W2    = (const float*)d_in[6];
    const float* b2    = (const float*)d_in[7];
    const float* W3    = (const float*)d_in[8];
    const float* b3    = (const float*)d_in[9];
    const float* Wo    = (const float*)d_in[10];
    const float* bo    = (const float*)d_in[11];
    const float* al    = (const float*)d_in[12];
    const float* dw    = (const float*)d_in[13];
    const float* db    = (const float*)d_in[14];
    float* out = (float*)d_out;

    char* ws = (char*)d_ws;
    float* sums = (float*)(ws + 0);           // 1024 f32
    float* ss   = (float*)(ws + 4096);        // 1024 f32 (scale | shift)
    int*   offs = (int*)  (ws + 8192);        // 32 int
    unsigned short* W1t = (unsigned short*)(ws + 16384);                       // 512x512
    unsigned short* W2t = (unsigned short*)(ws + 16384 + 524288);              // 256x512
    unsigned short* W3t = (unsigned short*)(ws + 16384 + 524288 + 262144);     // 256x256
    unsigned short* Xn  = (unsigned short*)(ws + ((size_t)1 << 20));           // 65536x512 bf16
    unsigned short* H1  = Xn + (size_t)N_NODES * IN_CH;                        // 65536x512 bf16
    unsigned short* H2  = H1 + (size_t)N_NODES * IN_CH;                        // 65536x256 bf16
    unsigned short* H3  = H2 + (size_t)N_NODES * HID;                          // 65536x256 bf16

    // BatchNorm statistics
    k_zero<<<4, 256, 0, stream>>>(sums, 1024);
    k_bn_partial<<<256, 256, 0, stream>>>(x, sums);
    k_bn_finalize<<<2, 256, 0, stream>>>(sums, gam, bet, ss);
    k_bn_apply_cvt<<<(N_NODES * IN_CH / 4) / 256, 256, 0, stream>>>(x, ss, Xn);

    // weight transpose + bf16 convert
    k_transpose_cvt<<<(IN_CH * 2 * HID + 255) / 256, 256, 0, stream>>>(W1, W1t, IN_CH, 2 * HID);
    k_transpose_cvt<<<(2 * HID * HID + 255) / 256, 256, 0, stream>>>(W2, W2t, 2 * HID, HID);
    k_transpose_cvt<<<(HID * HID + 255) / 256, 256, 0, stream>>>(W3, W3t, HID, HID);

    // GEMM chain (bf16 WMMA, f32 accumulate, async-LDS double-buffered B)
    dim3 g1(N_NODES / 64, (2 * HID) / 256);
    k_gemm_bf16<<<g1, 256, 0, stream>>>(Xn, W1t, b1, H1, IN_CH, 2 * HID, 1);
    dim3 g2(N_NODES / 64, HID / 256);
    k_gemm_bf16<<<g2, 256, 0, stream>>>(H1, W2t, b2, H2, 2 * HID, HID, 1);
    dim3 g3(N_NODES / 64, HID / 256);
    k_gemm_bf16<<<g3, 256, 0, stream>>>(H2, W3t, b3, H3, HID, HID, 0);

    // ragged scatter bookkeeping + padded-head output
    k_offsets<<<N_NODES / 256, 256, 0, stream>>>(batch, offs);
    k_fill_out<<<(MAX_LEN * NUM_GRAPHS * OUT_CH) / 256, 256, 0, stream>>>(out, bo, al, dw, db);
    k_head<<<N_NODES / 256, 256, 0, stream>>>(H3, Wo, bo, batch, offs, al, dw, db, out);
}
